// DNRI_MLP_Decoder_77378130805129
// MI455X (gfx1250) — compile-verified, matrix operand order checked by reading, not run on hardware
//
#include <hip/hip_runtime.h>

#define NV    64
#define ET    4
#define HID   128
#define INF   32
#define BATCH 64
#define NE    (NV * (NV - 1))   // 4032

typedef __attribute__((ext_vector_type(16))) _Float16 v16h;
typedef __attribute__((ext_vector_type(8)))  _Float16 v8h;
typedef __attribute__((ext_vector_type(8)))  float    v8f;

// Load one 16x32 (f16) WMMA operand fragment from a row-major f16 buffer.
// Per the CDNA5 ISA 16-bit A/B layouts: lanes 0-15 hold K={k0..k0+7, k0+16..k0+23},
// lanes 16-31 hold K={k0+8..k0+15, k0+24..k0+31}; row/col = lane%16 + rowBase.
static __device__ __forceinline__ v16h load_frag(const _Float16* __restrict__ base,
                                                 int rowBase, int strideH, int k0, int lane) {
  const int l  = lane & 15;
  const int kb = k0 + ((lane & 16) ? 8 : 0);
  const _Float16* p = base + (size_t)(rowBase + l) * strideH + kb;
  v8h lo = *(const v8h*)(p);
  v8h hi = *(const v8h*)(p + 16);
  v16h r;
#pragma unroll
  for (int i = 0; i < 8; ++i) { r[i] = lo[i]; r[i + 8] = hi[i]; }
  return r;
}

static __device__ __forceinline__ v8f wmma16(v16h a, v16h b, v8f c) {
  return __builtin_amdgcn_wmma_f32_16x16x32_f16(false, a, false, b, (short)0, c, false, false);
}

// ---------------------------------------------------------------- weight cvt
__global__ __launch_bounds__(256) void cvt_f16_kernel(const float* __restrict__ src,
                                                      _Float16* __restrict__ dst, int n) {
  int i = blockIdx.x * 256 + threadIdx.x;
  if (i < n) dst[i] = (_Float16)src[i];
}

// ------------------------------------------------- fused messages + aggregate
// One block per (batch b, receiver r). 63 incoming edges -> 64-row tile (row 63 zero).
__global__ __launch_bounds__(256) void msg_agg_kernel(
    const float* __restrict__ inputs, const float* __restrict__ edges,
    const _Float16* __restrict__ w1h, const float* __restrict__ b1,
    const _Float16* __restrict__ w2h, const float* __restrict__ b2,
    float* __restrict__ agg) {
  __shared__ alignas(16) _Float16 Xh[64 * 64];
  __shared__ alignas(16) _Float16 H1h[64 * HID];
  __shared__ float wE[64 * ET];

  const int blk  = blockIdx.x;
  const int b    = blk >> 6;
  const int r    = blk & 63;
  const int tid  = threadIdx.x;
  const int lane = tid & 31;
  const int wave = tid >> 5;
  const int hi   = (lane & 16) ? 8 : 0;
  const int l15  = lane & 15;
  const int n    = wave * 16 + l15;            // output column owned by this lane

  const float* inB = inputs + (size_t)b * NV * INF;

  // Build pre_msg tile: row m = [recv feats | sender s_m feats], s_m = m + (m>=r)
  for (int idx = tid; idx < 64 * 64; idx += 256) {
    int m = idx >> 6, c = idx & 63;
    float v = 0.f;
    if (m < 63) {
      int s = m + (m >= r ? 1 : 0);
      v = (c < INF) ? inB[r * INF + c] : inB[s * INF + (c - INF)];
    }
    Xh[idx] = (_Float16)v;
  }
  // Edge gates: e = s*63 + (r>s ? r-1 : r)
  if (tid < 64) {
    float w0 = 0.f, w1 = 0.f, w2 = 0.f, w3 = 0.f;
    if (tid < 63) {
      int s = tid + (tid >= r ? 1 : 0);
      int e = s * 63 + (r > s ? r - 1 : r);
      const float* ep = edges + ((size_t)b * NE + e) * ET;
      w0 = ep[0]; w1 = ep[1]; w2 = ep[2]; w3 = ep[3];
    }
    wE[tid * 4 + 0] = w0; wE[tid * 4 + 1] = w1;
    wE[tid * 4 + 2] = w2; wE[tid * 4 + 3] = w3;
  }
  __syncthreads();

  float accN = 0.f;

  for (int t = 1; t < ET; ++t) {   // SKIP_FIRST
    // ---- GEMM1: H1 = relu(X @ W1[t]^T + b1[t]),  M=64 K=64 N=128 ----
    const _Float16* W1t = w1h + (size_t)t * HID * 64;
    const float bias1 = b1[t * HID + n];
    v8f c0 = {}, c1 = {}, c2 = {}, c3 = {};
#pragma unroll
    for (int k = 0; k < 64; k += 32) {
      v16h bf = load_frag(W1t, wave * 16, 64, k, lane);
      v16h a0 = load_frag(Xh,  0, 64, k, lane);
      v16h a1 = load_frag(Xh, 16, 64, k, lane);
      v16h a2 = load_frag(Xh, 32, 64, k, lane);
      v16h a3 = load_frag(Xh, 48, 64, k, lane);
      c0 = wmma16(a0, bf, c0);
      c1 = wmma16(a1, bf, c1);
      c2 = wmma16(a2, bf, c2);
      c3 = wmma16(a3, bf, c3);
    }
#pragma unroll
    for (int i = 0; i < 8; ++i) {
      H1h[( 0 + i + hi) * HID + n] = (_Float16)fmaxf(c0[i] + bias1, 0.f);
      H1h[(16 + i + hi) * HID + n] = (_Float16)fmaxf(c1[i] + bias1, 0.f);
      H1h[(32 + i + hi) * HID + n] = (_Float16)fmaxf(c2[i] + bias1, 0.f);
      H1h[(48 + i + hi) * HID + n] = (_Float16)fmaxf(c3[i] + bias1, 0.f);
    }
    __syncthreads();

    // ---- GEMM2: M2 = relu(H1 @ W2[t]^T + b2[t]),  M=64 K=128 N=128 ----
    const _Float16* W2t = w2h + (size_t)t * HID * HID;
    const float bias2 = b2[t * HID + n];
    v8f d0 = {}, d1 = {}, d2 = {}, d3 = {};
#pragma unroll
    for (int k = 0; k < HID; k += 32) {
      v16h bf = load_frag(W2t, wave * 16, HID, k, lane);
      v16h a0 = load_frag(H1h,  0, HID, k, lane);
      v16h a1 = load_frag(H1h, 16, HID, k, lane);
      v16h a2 = load_frag(H1h, 32, HID, k, lane);
      v16h a3 = load_frag(H1h, 48, HID, k, lane);
      d0 = wmma16(a0, bf, d0);
      d1 = wmma16(a1, bf, d1);
      d2 = wmma16(a2, bf, d2);
      d3 = wmma16(a3, bf, d3);
    }
    // Gate by edge weight and reduce over rows into per-lane column partial.
#pragma unroll
    for (int i = 0; i < 8; ++i) {
      accN += fmaxf(d0[i] + bias2, 0.f) * wE[( 0 + i + hi) * 4 + t];
      accN += fmaxf(d1[i] + bias2, 0.f) * wE[(16 + i + hi) * 4 + t];
      accN += fmaxf(d2[i] + bias2, 0.f) * wE[(32 + i + hi) * 4 + t];
      accN += fmaxf(d3[i] + bias2, 0.f) * wE[(48 + i + hi) * 4 + t];
    }
    __syncthreads();   // protect H1h before next type overwrites it
  }

  // Lanes n and n+16 hold complementary row-halves of the same column.
  accN += __shfl_xor(accN, 16, 32);
  if (lane < 16) agg[((size_t)b * NV + r) * HID + n] = accN;
}

// ------------------------------------------------------- fused output MLP
// One block per 64 flattened (b,v) rows: 160 -> 128 -> 128 -> 32, out = in + mu.
__global__ __launch_bounds__(256) void out_mlp_kernel(
    const float* __restrict__ inputs, const float* __restrict__ agg,
    const _Float16* __restrict__ wo1, const float* __restrict__ bo1,
    const _Float16* __restrict__ wo2, const float* __restrict__ bo2,
    const _Float16* __restrict__ wmu, const float* __restrict__ bmu,
    float* __restrict__ out) {
  __shared__ alignas(16) _Float16 Xh[64 * 160];
  __shared__ alignas(16) _Float16 H1h[64 * HID];
  __shared__ alignas(16) _Float16 H2h[64 * HID];

  const int row0 = blockIdx.x * 64;
  const int tid  = threadIdx.x;
  const int lane = tid & 31;
  const int wave = tid >> 5;
  const int hi   = (lane & 16) ? 8 : 0;
  const int l15  = lane & 15;
  const int n    = wave * 16 + l15;

  for (int idx = tid; idx < 64 * 160; idx += 256) {
    int m = idx / 160, c = idx % 160;
    size_t row = (size_t)(row0 + m);
    float v = (c < INF) ? inputs[row * INF + c] : agg[row * HID + (c - INF)];
    Xh[idx] = (_Float16)v;
  }
  __syncthreads();

  // GEMM1: M=64 K=160 N=128
  {
    const float bias1 = bo1[n];
    v8f c0 = {}, c1 = {}, c2 = {}, c3 = {};
#pragma unroll
    for (int k = 0; k < 160; k += 32) {
      v16h bf = load_frag(wo1, wave * 16, 160, k, lane);
      v16h a0 = load_frag(Xh,  0, 160, k, lane);
      v16h a1 = load_frag(Xh, 16, 160, k, lane);
      v16h a2 = load_frag(Xh, 32, 160, k, lane);
      v16h a3 = load_frag(Xh, 48, 160, k, lane);
      c0 = wmma16(a0, bf, c0);
      c1 = wmma16(a1, bf, c1);
      c2 = wmma16(a2, bf, c2);
      c3 = wmma16(a3, bf, c3);
    }
#pragma unroll
    for (int i = 0; i < 8; ++i) {
      H1h[( 0 + i + hi) * HID + n] = (_Float16)fmaxf(c0[i] + bias1, 0.f);
      H1h[(16 + i + hi) * HID + n] = (_Float16)fmaxf(c1[i] + bias1, 0.f);
      H1h[(32 + i + hi) * HID + n] = (_Float16)fmaxf(c2[i] + bias1, 0.f);
      H1h[(48 + i + hi) * HID + n] = (_Float16)fmaxf(c3[i] + bias1, 0.f);
    }
  }
  __syncthreads();

  // GEMM2: M=64 K=128 N=128
  {
    const float bias2 = bo2[n];
    v8f d0 = {}, d1 = {}, d2 = {}, d3 = {};
#pragma unroll
    for (int k = 0; k < HID; k += 32) {
      v16h bf = load_frag(wo2, wave * 16, HID, k, lane);
      v16h a0 = load_frag(H1h,  0, HID, k, lane);
      v16h a1 = load_frag(H1h, 16, HID, k, lane);
      v16h a2 = load_frag(H1h, 32, HID, k, lane);
      v16h a3 = load_frag(H1h, 48, HID, k, lane);
      d0 = wmma16(a0, bf, d0);
      d1 = wmma16(a1, bf, d1);
      d2 = wmma16(a2, bf, d2);
      d3 = wmma16(a3, bf, d3);
    }
#pragma unroll
    for (int i = 0; i < 8; ++i) {
      H2h[( 0 + i + hi) * HID + n] = (_Float16)fmaxf(d0[i] + bias2, 0.f);
      H2h[(16 + i + hi) * HID + n] = (_Float16)fmaxf(d1[i] + bias2, 0.f);
      H2h[(32 + i + hi) * HID + n] = (_Float16)fmaxf(d2[i] + bias2, 0.f);
      H2h[(48 + i + hi) * HID + n] = (_Float16)fmaxf(d3[i] + bias2, 0.f);
    }
  }
  __syncthreads();

  // GEMM3 (mu): M=64 K=128 N=32 -> 2 Ntiles x 4 Mtiles, one tile per wave.
  {
    const int mt = wave & 3;
    const int nt = wave >> 2;
    const int n3 = nt * 16 + l15;
    const float bm = bmu[n3];
    v8f e = {};
#pragma unroll
    for (int k = 0; k < HID; k += 32) {
      v16h bf = load_frag(wmu, nt * 16, HID, k, lane);
      v16h a  = load_frag(H2h, mt * 16, HID, k, lane);
      e = wmma16(a, bf, e);
    }
#pragma unroll
    for (int i = 0; i < 8; ++i) {
      size_t row = (size_t)(row0 + mt * 16 + i + hi);
      out[row * INF + n3] = inputs[row * INF + n3] + e[i] + bm;
    }
  }
}

// ----------------------------------------------------------------- launcher
extern "C" void kernel_launch(void* const* d_in, const int* in_sizes, int n_in,
                              void* d_out, int out_size, void* d_ws, size_t ws_size,
                              hipStream_t stream) {
  const float* inputs = (const float*)d_in[0];
  const float* edges  = (const float*)d_in[1];
  const float* m1w    = (const float*)d_in[2];
  const float* m1b    = (const float*)d_in[3];
  const float* m2w    = (const float*)d_in[4];
  const float* m2b    = (const float*)d_in[5];
  const float* o1w    = (const float*)d_in[6];
  const float* o1b    = (const float*)d_in[7];
  const float* o2w    = (const float*)d_in[8];
  const float* o2b    = (const float*)d_in[9];
  const float* muw    = (const float*)d_in[10];
  const float* mub    = (const float*)d_in[11];
  float* out = (float*)d_out;

  const int N1 = ET * HID * 64;          // msg_fc1_w   32768
  const int N2 = ET * HID * HID;         // msg_fc2_w   65536
  const int N3 = HID * (INF + HID);      // out_fc1_w   20480
  const int N4 = HID * HID;              // out_fc2_w   16384
  const int N5 = INF * HID;              // mu_w         4096

  char* ws = (char*)d_ws;
  _Float16* w1h = (_Float16*)ws;
  _Float16* w2h = w1h + N1;
  _Float16* wo1 = w2h + N2;
  _Float16* wo2 = wo1 + N3;
  _Float16* wmu = wo2 + N4;
  size_t aggOff = (((size_t)(N1 + N2 + N3 + N4 + N5) * 2) + 255) & ~(size_t)255;
  float* agg = (float*)(ws + aggOff);    // BATCH*NV*HID f32 = 2 MB

  cvt_f16_kernel<<<(N1 + 255) / 256, 256, 0, stream>>>(m1w, w1h, N1);
  cvt_f16_kernel<<<(N2 + 255) / 256, 256, 0, stream>>>(m2w, w2h, N2);
  cvt_f16_kernel<<<(N3 + 255) / 256, 256, 0, stream>>>(o1w, wo1, N3);
  cvt_f16_kernel<<<(N4 + 255) / 256, 256, 0, stream>>>(o2w, wo2, N4);
  cvt_f16_kernel<<<(N5 + 255) / 256, 256, 0, stream>>>(muw, wmu, N5);

  msg_agg_kernel<<<BATCH * NV, 256, 0, stream>>>(inputs, edges, w1h, m1b, w2h, m2b, agg);

  out_mlp_kernel<<<BATCH * NV / 64, 256, 0, stream>>>(inputs, agg, wo1, o1b, wo2, o2b,
                                                      wmu, mub, out);
}
